// MS_MoE_Conv_7301444403349
// MI455X (gfx1250) — matrix operand order repaired and checked by reading, not compiled, not run on hardware
//
#include <hip/hip_runtime.h>
#include <cstdint>

typedef _Float16 v16h __attribute__((ext_vector_type(16)));
typedef _Float16 v8h  __attribute__((ext_vector_type(8)));
typedef float    v8f  __attribute__((ext_vector_type(8)));

#define T_    4
#define B_    32
#define C_    384
#define HF_   1536
#define HW_   196
#define E_    8
#define N_    128            // T*B tokens (routing granularity)
#define NPAIR 256            // N * TOP_K
#define MPAD  208            // 196 padded to 13 WMMA M-tiles
#define MT_   13
#define PADK  392            // C_ + 8 f16 pad -> conflict-free ds_read_b128
#define SROW  (HF_ / 8)      // 192 bytes: one bit-packed s2 row
#define OUT_ELEMS (T_*B_*C_*HW_)   // 9,633,792 ; aux scalar appended after

// ---------------------------------------------------------------- weights->f16
__global__ __launch_bounds__(256) void wcvt_kernel(
    const float* __restrict__ w1, const float* __restrict__ w2,
    _Float16* __restrict__ w1h, _Float16* __restrict__ w2h) {
  const int n1 = E_ * HF_ * C_;
  for (int i = blockIdx.x * 256 + threadIdx.x; i < 2 * n1; i += gridDim.x * 256) {
    if (i < n1) w1h[i] = (_Float16)w1[i];
    else        w2h[i - n1] = (_Float16)w2[i - n1];
  }
}

// ------------------------------------------------- fold bias+BN into scale/off
__global__ __launch_bounds__(256) void scales_kernel(
    const float* __restrict__ b1, const float* __restrict__ g1,
    const float* __restrict__ be1, const float* __restrict__ m1,
    const float* __restrict__ v1,
    const float* __restrict__ b2, const float* __restrict__ g2,
    const float* __restrict__ be2, const float* __restrict__ m2,
    const float* __restrict__ v2,
    float* __restrict__ scale1, float* __restrict__ off1,
    float* __restrict__ scale2, float* __restrict__ off2) {
  int i = blockIdx.x * 256 + threadIdx.x;
  const int n1 = E_ * HF_;
  if (i < n1) {
    float inv = g1[i] * rsqrtf(v1[i] + 1e-5f);
    scale1[i] = inv;
    off1[i]   = b1[i] * inv + (be1[i] - m1[i] * inv);
  } else if (i < n1 + E_ * C_) {
    int j = i - n1;
    float inv = g2[j] * rsqrtf(v2[j] + 1e-5f);
    scale2[j] = inv;
    off2[j]   = b2[j] * inv + (be2[j] - m2[j] * inv);
  }
}

// ---------------------- LIF multistep (tau=2) + spatial mean of spikes (N,C)
__global__ __launch_bounds__(256) void lif_mean_kernel(
    const float* __restrict__ x, float* __restrict__ ms) {
  __shared__ float red[256];
  const int bc = blockIdx.x;               // over B_*C_
  const int b = bc / C_, c = bc % C_;
  const int tid = threadIdx.x;
  float v = 0.f;
  for (int t = 0; t < T_; ++t) {
    float s = 0.f;
    if (tid < HW_) {
      float xv = x[(((size_t)(t * B_ + b)) * C_ + c) * HW_ + tid];
      v += (xv - v) * 0.5f;                // decay_input, tau=2
      s = (v >= 1.f) ? 1.f : 0.f;          // v_threshold = 1
      v *= (1.f - s);                      // hard reset
    }
    red[tid] = s;
    __syncthreads();
    #pragma unroll
    for (int st = 128; st > 0; st >>= 1) {
      if (tid < st) red[tid] += red[tid + st];
      __syncthreads();
    }
    if (tid == 0) ms[(size_t)(t * B_ + b) * C_ + c] = red[0] * (1.f / (float)HW_);
    __syncthreads();
  }
}

// -------- router: logits = BN(ms @ rw^T + rb), softmax, top-2, gate, aux loss
__global__ __launch_bounds__(128) void router_kernel(
    const float* __restrict__ ms, const float* __restrict__ rw,
    const float* __restrict__ rb, const float* __restrict__ rg,
    const float* __restrict__ rbeta, const float* __restrict__ rmean,
    const float* __restrict__ rvar,
    int* __restrict__ tke, float* __restrict__ gatew, float* __restrict__ out) {
  __shared__ float cnt[E_], psum[E_];
  const int n = threadIdx.x;               // one token per thread, 128 threads
  if (n < E_) { cnt[n] = 0.f; psum[n] = 0.f; }
  __syncthreads();
  float probs[E_];
  float mx = -1e30f;
  #pragma unroll
  for (int e = 0; e < E_; ++e) {
    float acc = rb[e];
    const float* msr = ms + (size_t)n * C_;
    const float* rwr = rw + (size_t)e * C_;
    for (int c = 0; c < C_; ++c) acc += msr[c] * rwr[c];
    float inv = rg[e] * rsqrtf(rvar[e] + 1e-5f);
    probs[e] = acc * inv + (rbeta[e] - rmean[e] * inv);   // logits for now
    mx = fmaxf(mx, probs[e]);
  }
  float sum = 0.f;
  #pragma unroll
  for (int e = 0; e < E_; ++e) { probs[e] = __expf(probs[e] - mx); sum += probs[e]; }
  float rs = 1.f / sum;
  #pragma unroll
  for (int e = 0; e < E_; ++e) probs[e] *= rs;
  int i0 = 0; float p0 = probs[0];
  #pragma unroll
  for (int e = 1; e < E_; ++e) if (probs[e] > p0) { p0 = probs[e]; i0 = e; }
  int i1 = (i0 == 0) ? 1 : 0; float p1 = probs[i1];
  #pragma unroll
  for (int e = 0; e < E_; ++e)
    if (e != i0 && probs[e] > p1) { p1 = probs[e]; i1 = e; }
  float norm = 1.f / (p0 + p1);
  tke[n * 2 + 0] = i0; gatew[n * 2 + 0] = p0 * norm;
  tke[n * 2 + 1] = i1; gatew[n * 2 + 1] = p1 * norm;
  #pragma unroll
  for (int e = 0; e < E_; ++e) atomicAdd(&psum[e], probs[e]);
  atomicAdd(&cnt[i0], 1.f);
  atomicAdd(&cnt[i1], 1.f);
  __syncthreads();
  if (n == 0) {
    float aux = 0.f;
    #pragma unroll
    for (int e = 0; e < E_; ++e)
      aux += (cnt[e] * (1.f / N_)) * (psum[e] * (1.f / N_));
    out[OUT_ELEMS] = 0.01f * (float)E_ * aux;   // AUX_W * E * sum(f*p)
  }
}

// ------------- layer 1: s1(x>=tau) @ W1 -> BN -> spike -> bit-packed s2 in L2
// grid: (pair, hf_chunk of 256); block 256 = 8 waves, 2 N-tiles per wave.
// w1h is [n][k] row-major == contiguous-k B-fragment per-lane layout, so the
// compiler hoists all 24 B fragments into VGPRs outside the mt loop.
__global__ __launch_bounds__(256) void expert_l1_kernel(
    const float* __restrict__ x, const _Float16* __restrict__ w1h,
    const float* __restrict__ scale1, const float* __restrict__ off1,
    const int* __restrict__ tke, const float* __restrict__ taus,
    unsigned char* __restrict__ s2) {
  __shared__ _Float16 sA[16 * PADK];       // one transposed spike strip (12.5 KB)
  const int pair = blockIdx.x;
  const int chunk = blockIdx.y;            // 0..5  (6 * 256 = HF_)
  const int n = pair >> 1;
  const int e = tke[pair];
  const float tau = taus[e];
  const int tid = threadIdx.x, lane = tid & 31, wid = tid >> 5;
  const float* xn = x + (size_t)n * C_ * HW_;
  const _Float16* w1e = w1h + (size_t)e * HF_ * C_;
  const int obase0 = chunk * 256 + wid * 32;     // wave's two 16-wide hf tiles
  const int obase1 = obase0 + 16;
  const int o0 = obase0 + (lane & 15);
  const int o1 = obase1 + (lane & 15);
  const int arow = (lane & 15) * PADK;
  const int aklo = (lane < 16) ? 0 : 16;         // contiguous A-frag K pattern
  const int bklo = (lane < 16) ? 0 : 16;         // contiguous B-frag K pattern (matches A)
  const float sc0 = scale1[e * HF_ + o0], of0 = off1[e * HF_ + o0];
  const float sc1 = scale1[e * HF_ + o1], of1 = off1[e * HF_ + o1];
  unsigned char* s2p = s2 + (size_t)pair * MPAD * SROW;

  for (int mt = 0; mt < MT_; ++mt) {
    __syncthreads();                       // strip reuse fence
    // stage s1 strip, transposed: sA[p][c] = (x[c][p] >= tau)
    for (int i = tid; i < 16 * C_; i += 256) {
      int pr = i & 15, c = i >> 4;
      int p = mt * 16 + pr;
      float xv = (p < HW_) ? xn[(size_t)c * HW_ + p] : 0.f;
      sA[pr * PADK + c] = (_Float16)((xv >= tau) ? 1.f : 0.f);
    }
    __syncthreads();

    v8f acc0 = {}; v8f acc1 = {};
    #pragma unroll
    for (int kt = 0; kt < C_ / 32; ++kt) {
      const int kb = kt * 32;
      v8h alo = *(const v8h*)&sA[arow + kb + aklo];
      v8h ahi = *(const v8h*)&sA[arow + kb + aklo + 8];
      v16h af = __builtin_shufflevector(alo, ahi, 0, 1, 2, 3, 4, 5, 6, 7,
                                                  8, 9, 10, 11, 12, 13, 14, 15);
      v16h bf0 = *(const v16h*)(w1e + (size_t)o0 * C_ + kb + bklo);
      v16h bf1 = *(const v16h*)(w1e + (size_t)o1 * C_ + kb + bklo);
      acc0 = __builtin_amdgcn_wmma_f32_16x16x32_f16(false, af, false, bf0,
                                                    (short)0, acc0, false, false);
      acc1 = __builtin_amdgcn_wmma_f32_16x16x32_f16(false, af, false, bf1,
                                                    (short)0, acc1, false, false);
    }

    // epilogue: spike(BN(h)/tau - 1) = (h*scale+off >= tau); ballot-pack bits
    #pragma unroll
    for (int j = 0; j < 8; ++j) {
      int m = mt * 16 + ((lane < 16) ? j : j + 8);
      bool pr0 = (m < HW_) && (acc0[j] * sc0 + of0 >= tau);
      bool pr1 = (m < HW_) && (acc1[j] * sc1 + of1 >= tau);
      unsigned pm0 = __builtin_amdgcn_ballot_w32(pr0);
      unsigned pm1 = __builtin_amdgcn_ballot_w32(pr1);
      if (lane == 0) {
        int mA = mt * 16 + j, mB = mA + 8;   // low/high 16 lanes = rows j / j+8
        *(uint16_t*)(s2p + (size_t)mA * SROW + (obase0 >> 3)) = (uint16_t)(pm0 & 0xffff);
        *(uint16_t*)(s2p + (size_t)mB * SROW + (obase0 >> 3)) = (uint16_t)(pm0 >> 16);
        *(uint16_t*)(s2p + (size_t)mA * SROW + (obase1 >> 3)) = (uint16_t)(pm1 & 0xffff);
        *(uint16_t*)(s2p + (size_t)mB * SROW + (obase1 >> 3)) = (uint16_t)(pm1 >> 16);
      }
    }
  }
}

// ------- layer 2: s2 @ W2 -> BN, gate-combine both experts, +residual, store
// grid: n (128); block 256 = 8 waves, 3 C-tiles per wave (24 tiles = 384).
// A comes from the 10 MB bit-packed s2 (L2-resident), B streams w2h from L2.
__global__ __launch_bounds__(256) void expert_l2_kernel(
    const float* __restrict__ x, const _Float16* __restrict__ w2h,
    const unsigned char* __restrict__ s2,
    const float* __restrict__ scale2, const float* __restrict__ off2,
    const int* __restrict__ tke, const float* __restrict__ gatew,
    float* __restrict__ out) {
  const int n = blockIdx.x;
  const int tid = threadIdx.x, lane = tid & 31, wid = tid >> 5;
  const int ct0 = wid, ct1 = wid + 8, ct2 = wid + 16;
  const int c0 = ct0 * 16 + (lane & 15);
  const int c1 = ct1 * 16 + (lane & 15);
  const int c2 = ct2 * 16 + (lane & 15);
  const int ash = (lane < 16) ? 0 : 16;    // A-frag bit offset in packed word
  const int bklo = (lane < 16) ? 0 : 16;

  for (int mt = 0; mt < MT_; ++mt) {
    float tot0[8], tot1[8], tot2[8];
    #pragma unroll
    for (int j = 0; j < 8; ++j) { tot0[j] = 0.f; tot1[j] = 0.f; tot2[j] = 0.f; }

    #pragma unroll
    for (int kk = 0; kk < 2; ++kk) {       // both routed experts for token n
      const int pair = n * 2 + kk;
      const int e = tke[pair];
      const float g = gatew[pair];
      const unsigned* arow = (const unsigned*)(s2 + (size_t)pair * MPAD * SROW +
                                               (size_t)(mt * 16 + (lane & 15)) * SROW);
      const _Float16* w2e = w2h + (size_t)e * C_ * HF_;
      v8f a0 = {}, a1 = {}, a2 = {};
      #pragma unroll 4
      for (int kt = 0; kt < HF_ / 32; ++kt) {
        const int kb = kt * 32;
        unsigned bits = arow[kt];          // K = kb..kb+31 spike bits (L2-hot)
        v16h bf0 = *(const v16h*)(w2e + (size_t)c0 * HF_ + kb + bklo);
        v16h bf1 = *(const v16h*)(w2e + (size_t)c1 * HF_ + kb + bklo);
        v16h bf2 = *(const v16h*)(w2e + (size_t)c2 * HF_ + kb + bklo);
        v16h af;
        #pragma unroll
        for (int q = 0; q < 8; ++q) af[q]     = (_Float16)((bits >> (ash + q)) & 1u);
        #pragma unroll
        for (int q = 0; q < 8; ++q) af[q + 8] = (_Float16)((bits >> (ash + 8 + q)) & 1u);
        a0 = __builtin_amdgcn_wmma_f32_16x16x32_f16(false, af, false, bf0,
                                                    (short)0, a0, false, false);
        a1 = __builtin_amdgcn_wmma_f32_16x16x32_f16(false, af, false, bf1,
                                                    (short)0, a1, false, false);
        a2 = __builtin_amdgcn_wmma_f32_16x16x32_f16(false, af, false, bf2,
                                                    (short)0, a2, false, false);
      }
      const float s0 = scale2[e * C_ + c0], f0 = off2[e * C_ + c0];
      const float s1 = scale2[e * C_ + c1], f1 = off2[e * C_ + c1];
      const float s2c = scale2[e * C_ + c2], f2 = off2[e * C_ + c2];
      #pragma unroll
      for (int j = 0; j < 8; ++j) {
        tot0[j] += g * (a0[j] * s0 + f0);
        tot1[j] += g * (a1[j] * s1 + f1);
        tot2[j] += g * (a2[j] * s2c + f2);
      }
    }

    // out = x (residual; gates sum to 1) + sum_e gate*o
    #pragma unroll
    for (int j = 0; j < 8; ++j) {
      int p = mt * 16 + ((lane < 16) ? j : j + 8);
      if (p < HW_) {
        size_t i0 = ((size_t)n * C_ + c0) * HW_ + p;
        size_t i1 = ((size_t)n * C_ + c1) * HW_ + p;
        size_t i2 = ((size_t)n * C_ + c2) * HW_ + p;
        out[i0] = x[i0] + tot0[j];
        out[i1] = x[i1] + tot1[j];
        out[i2] = x[i2] + tot2[j];
      }
    }
  }
}

// --------------------------------------------------------------------- launch
extern "C" void kernel_launch(void* const* d_in, const int* in_sizes, int n_in,
                              void* d_out, int out_size, void* d_ws, size_t ws_size,
                              hipStream_t stream) {
  (void)in_sizes; (void)n_in; (void)out_size; (void)ws_size;
  const float* x     = (const float*)d_in[0];
  const float* rw    = (const float*)d_in[1];
  const float* rb    = (const float*)d_in[2];
  const float* rg    = (const float*)d_in[3];
  const float* rbeta = (const float*)d_in[4];
  const float* rmean = (const float*)d_in[5];
  const float* rvar  = (const float*)d_in[6];
  const float* w1    = (const float*)d_in[7];
  const float* b1    = (const float*)d_in[8];
  const float* g1    = (const float*)d_in[9];
  const float* be1   = (const float*)d_in[10];
  const float* m1    = (const float*)d_in[11];
  const float* v1    = (const float*)d_in[12];
  const float* w2    = (const float*)d_in[13];
  const float* b2    = (const float*)d_in[14];
  const float* g2    = (const float*)d_in[15];
  const float* be2   = (const float*)d_in[16];
  const float* m2    = (const float*)d_in[17];
  const float* v2    = (const float*)d_in[18];
  const float* taus  = (const float*)d_in[19];
  float* out = (float*)d_out;

  // workspace carve (256B aligned slabs), ~29.4 MB total
  uint8_t* w = (uint8_t*)d_ws;
  size_t off = 0;
  auto carve = [&](size_t bytes) -> void* {
    void* p = w + off;
    off += (bytes + 255) & ~(size_t)255;
    return p;
  };
  _Float16* w1h   = (_Float16*)carve((size_t)E_ * HF_ * C_ * 2);
  _Float16* w2h   = (_Float16*)carve((size_t)E_ * C_ * HF_ * 2);
  float* scale1   = (float*)carve((size_t)E_ * HF_ * 4);
  float* off1     = (float*)carve((size_t)E_ * HF_ * 4);
  float* scale2   = (float*)carve((size_t)E_ * C_ * 4);
  float* off2     = (float*)carve((size_t)E_ * C_ * 4);
  float* ms       = (float*)carve((size_t)N_ * C_ * 4);
  int*   tke      = (int*)carve((size_t)NPAIR * 4);
  float* gw       = (float*)carve((size_t)NPAIR * 4);
  unsigned char* s2bits = (unsigned char*)carve((size_t)NPAIR * MPAD * SROW);

  wcvt_kernel<<<4096, 256, 0, stream>>>(w1, w2, w1h, w2h);
  scales_kernel<<<(E_ * HF_ + E_ * C_ + 255) / 256, 256, 0, stream>>>(
      b1, g1, be1, m1, v1, b2, g2, be2, m2, v2, scale1, off1, scale2, off2);
  lif_mean_kernel<<<B_ * C_, 256, 0, stream>>>(x, ms);
  router_kernel<<<1, 128, 0, stream>>>(ms, rw, rb, rg, rbeta, rmean, rvar,
                                       tke, gw, out);
  expert_l1_kernel<<<dim3(NPAIR, HF_ / 256), 256, 0, stream>>>(
      x, w1h, scale1, off1, tke, taus, s2bits);
  expert_l2_kernel<<<N_, 256, 0, stream>>>(
      x, w2h, s2bits, scale2, off2, tke, gw, out);
}
